// MDTA_9526237462868
// MI455X (gfx1250) — matrix operand order repaired and checked by reading, not compiled, not run on hardware
//
#include <hip/hip_runtime.h>

// ---------------------------------------------------------------------------
// MDTA (Restormer transposed attention) for gfx1250, bf16 WMMA pipeline with
// GLOBAL_LOAD_ASYNC_TO_LDS staging. B=8, C=192, HEADS=4 (c=48), NPIX=16384.
// ---------------------------------------------------------------------------

typedef __attribute__((ext_vector_type(16))) __bf16 v16bf;
typedef __attribute__((ext_vector_type(8)))  float  v8f;

#define NPIX   16384
#define CDIM   192
#define C3     576
#define BATCH  8
#define HEADS  4
#define HC     48
#define EPSN   1e-12f

// Hardware f32->bf16 (fptrunc = round-to-nearest-even, matches jnp cast)
__device__ __forceinline__ __bf16 f2bf(float f) { return (__bf16)f; }

// Low 32 bits of a generic pointer into LDS == LDS byte offset (ISA 10.2).
__device__ __forceinline__ unsigned lds_off(const void* p) {
  return (unsigned)(uintptr_t)p;
}

// One lane copies 16B global -> LDS asynchronously (ASYNCcnt).
__device__ __forceinline__ void async_copy16(unsigned lds, const void* gptr) {
  asm volatile("global_load_async_to_lds_b128 %0, %1, off"
               :: "v"(lds), "v"(gptr) : "memory");
}
__device__ __forceinline__ void async_wait_all() {
  asm volatile("s_wait_asynccnt 0" ::: "memory");
}

// ---------------------------------------------------------------------------
// Prep: convert weight matrices to bf16 once (reused by thousands of blocks).
// ---------------------------------------------------------------------------
__global__ __launch_bounds__(256) void cvt_w_kernel(
    const float* __restrict__ src, __bf16* __restrict__ dst, int n)
{
  int i = blockIdx.x * 256 + threadIdx.x;
  if (i < n) dst[i] = f2bf(src[i]);
}

// ---------------------------------------------------------------------------
// Kernel 1/6: pointwise 1x1 conv as GEMM.  Y[b,m,p] = sum_k W[m,k]*X[b,k,p]
// Block = 128 threads (4 waves) -> 64 output rows x 16 pixels.
// f32 activation tile async-staged to LDS, converted once to a transposed
// bf16 tile [pixel][k] so each lane's B-fragment read is contiguous.
// ---------------------------------------------------------------------------
__global__ __launch_bounds__(128) void pw_gemm_kernel(
    const __bf16* __restrict__ W,  // [M][K] bf16
    const float* __restrict__ X,   // [B][K][NPIX] f32
    float* __restrict__ Y,         // [B][M][NPIX] f32
    int M, int K)
{
  __shared__ float  tileF[CDIM][20];   // staged f32, row stride 80B (16B aligned)
  __shared__ __bf16 tileB[16][200];    // bf16 [pixel][k]

  const int p0 = blockIdx.x * 16;
  const int m0b = blockIdx.y * 64;
  const int b = blockIdx.z;
  const float* Xb = X + (size_t)b * K * NPIX + p0;

  // Async stage-in: K*16 floats as K*4 16B chunks (exactly 6 per thread).
  for (int f = threadIdx.x; f < K * 4; f += 128) {
    int c = f >> 2, g = (f & 3) * 4;
    async_copy16(lds_off(&tileF[c][g]), Xb + (size_t)c * NPIX + g);
  }
  async_wait_all();
  __syncthreads();

  // Convert + transpose to bf16 once.
  for (int i = threadIdx.x; i < K * 16; i += 128) {
    int c = i >> 4, n = i & 15;
    tileB[n][c] = f2bf(tileF[c][n]);
  }
  __syncthreads();

  const int wave = threadIdx.x >> 5;
  const int lane = threadIdx.x & 31;
  const int m0 = m0b + wave * 16;
  const int n = lane & 15;
  const int kbA = (lane < 16) ? 0 : 16;   // A: contiguous k = 16h + s (matches B)
  const int kbB = (lane < 16) ? 0 : 16;   // B 32x16 fragment layout
  const __bf16* Wr = W + (size_t)(m0 + n) * K;

  v8f acc = {};
  for (int k0 = 0; k0 < K; k0 += 32) {
    v16bf a, bb;
#pragma unroll
    for (int j = 0; j < 8; ++j) {
      a[j]     = Wr[k0 + kbA + j];
      a[j + 8] = Wr[k0 + kbA + 8 + j];
    }
#pragma unroll
    for (int j = 0; j < 16; ++j)
      bb[j] = tileB[n][k0 + kbB + j];
    acc = __builtin_amdgcn_wmma_f32_16x16x32_bf16(false, a, false, bb,
                                                  (short)0, acc, false, false);
  }

  const int hi = lane >> 4;
  float* Yb = Y + (size_t)b * M * NPIX;
#pragma unroll
  for (int r = 0; r < 8; ++r) {
    int orow = m0 + r + 8 * hi;               // C/D: VGPR r -> M = r + 8*hi
    Yb[(size_t)orow * NPIX + p0 + n] = acc[r];
  }
}

// ---------------------------------------------------------------------------
// Kernel 2: depthwise 3x3 conv (pad 1) f32 -> bf16, fused sum-of-squares
// partials for the q/k L2 norms (channels < 384). Deterministic partials.
// ---------------------------------------------------------------------------
__global__ __launch_bounds__(256) void dwconv_kernel(
    const float* __restrict__ Q0,      // [B][576][128][128] f32
    const float* __restrict__ Wdw,     // [576][9]
    __bf16* __restrict__ Q1,           // [B][576][16384] bf16
    float* __restrict__ sumsq_part)    // [B*384][64]
{
  __shared__ float tile[18][19];
  __shared__ float wsum[8];

  const int bx = blockIdx.x, by = blockIdx.y;
  const int zc = blockIdx.z;                 // b*576 + ch
  const int b = zc / C3, ch = zc % C3;
  const int x0 = bx * 16, y0 = by * 16;
  const float* src = Q0 + (size_t)zc * NPIX;

  for (int i = threadIdx.x; i < 18 * 18; i += 256) {
    int ly = i / 18, lx = i % 18;
    int gy = y0 + ly - 1, gx = x0 + lx - 1;
    float v = 0.f;
    if (gy >= 0 && gy < 128 && gx >= 0 && gx < 128) v = src[gy * 128 + gx];
    tile[ly][lx] = v;
  }
  __syncthreads();

  const float* wc = Wdw + ch * 9;            // uniform -> scalar loads
  const int tx = threadIdx.x & 15, ty = threadIdx.x >> 4;
  float s = wc[0] * tile[ty + 0][tx + 0] + wc[1] * tile[ty + 0][tx + 1] + wc[2] * tile[ty + 0][tx + 2]
          + wc[3] * tile[ty + 1][tx + 0] + wc[4] * tile[ty + 1][tx + 1] + wc[5] * tile[ty + 1][tx + 2]
          + wc[6] * tile[ty + 2][tx + 0] + wc[7] * tile[ty + 2][tx + 1] + wc[8] * tile[ty + 2][tx + 2];

  Q1[(size_t)zc * NPIX + (size_t)(y0 + ty) * 128 + x0 + tx] = f2bf(s);

  float v2 = s * s;
#pragma unroll
  for (int off = 16; off > 0; off >>= 1) v2 += __shfl_xor(v2, off, 32);
  if ((threadIdx.x & 31) == 0) wsum[threadIdx.x >> 5] = v2;
  __syncthreads();
  if (threadIdx.x == 0 && ch < 2 * CDIM) {
    float t = 0.f;
#pragma unroll
    for (int i = 0; i < 8; ++i) t += wsum[i];
    sumsq_part[((size_t)b * 384 + ch) * 64 + by * 8 + bx] = t;
  }
}

__global__ void reduce_sumsq_kernel(const float* __restrict__ part,
                                    float* __restrict__ sumsq)
{
  int i = blockIdx.x * 256 + threadIdx.x;
  if (i < BATCH * 384) {
    float s = 0.f;
    for (int j = 0; j < 64; ++j) s += part[(size_t)i * 64 + j];
    sumsq[i] = s;
  }
}

// ---------------------------------------------------------------------------
// Kernel 3: Gram matrix partials. attn_raw[b,h,c,d] = sum_n q[c,n]*k[d,n].
// One wave per 16x16 output tile, K(=pixels) split 16 ways -> deterministic.
// ---------------------------------------------------------------------------
__global__ __launch_bounds__(32) void gram_kernel(
    const __bf16* __restrict__ Q1,     // [B][576][NPIX]
    float* __restrict__ attn_part)     // [16][B*4][48][48]
{
  const int tile = blockIdx.x;          // 0..8
  const int mi = tile / 3, di = tile % 3;
  const int bh = blockIdx.y;            // 0..31
  const int b = bh >> 2, head = bh & 3;
  const int kc = blockIdx.z;            // 0..15 pixel chunk (1024 each)
  const int lane = threadIdx.x;

  const __bf16* qrow = Q1 + ((size_t)b * C3 + head * HC + mi * 16 + (lane & 15)) * NPIX;
  const __bf16* krow = Q1 + ((size_t)b * C3 + CDIM + head * HC + di * 16 + (lane & 15)) * NPIX;
  const int kbA = (lane < 16) ? 0 : 16;
  const int kbB = (lane < 16) ? 0 : 16;

  v8f acc = {};
  int n0 = kc * 1024;
  for (int it = 0; it < 32; ++it, n0 += 32) {
    __builtin_prefetch(qrow + n0 + 512, 0, 0);   // global_prefetch_b8
    __builtin_prefetch(krow + n0 + 512, 0, 0);
    v16bf a, bb;
#pragma unroll
    for (int j = 0; j < 8; ++j) {
      a[j]     = qrow[n0 + kbA + j];
      a[j + 8] = qrow[n0 + kbA + 8 + j];
    }
#pragma unroll
    for (int j = 0; j < 16; ++j)
      bb[j] = krow[n0 + kbB + j];
    acc = __builtin_amdgcn_wmma_f32_16x16x32_bf16(false, a, false, bb,
                                                  (short)0, acc, false, false);
  }

  const int hi = lane >> 4;
  float* out = attn_part + ((size_t)kc * 32 + bh) * (HC * HC);
#pragma unroll
  for (int r = 0; r < 8; ++r) {
    int rm = mi * 16 + r + 8 * hi;
    out[rm * HC + di * 16 + (lane & 15)] = acc[r];
  }
}

// ---------------------------------------------------------------------------
// Kernel 4: reduce Gram partials, fold 1/||q||, 1/||k||, temperature,
// softmax over d (48). One thread per attention row.
// ---------------------------------------------------------------------------
__global__ __launch_bounds__(64) void softmax_kernel(
    const float* __restrict__ attn_part,  // [16][32][48][48]
    const float* __restrict__ sumsq,      // [B][384]
    const float* __restrict__ temperature,// [4]
    float* __restrict__ attn)             // [32][48][48]
{
  const int bh = blockIdx.x;
  const int b = bh >> 2, head = bh & 3;
  const int m = threadIdx.x;
  if (m >= HC) return;

  const float qs = 1.f / fmaxf(sqrtf(sumsq[b * 384 + head * HC + m]), EPSN);
  const float tmp = temperature[head];

  float vals[HC];
  float mx = -3.4e38f;
  for (int d = 0; d < HC; ++d) {
    float s = 0.f;
    for (int j = 0; j < 16; ++j)
      s += attn_part[((size_t)j * 32 + bh) * (HC * HC) + m * HC + d];
    float ks = 1.f / fmaxf(sqrtf(sumsq[b * 384 + CDIM + head * HC + d]), EPSN);
    float v = s * qs * ks * tmp;
    vals[d] = v;
    mx = fmaxf(mx, v);
  }
  float sum = 0.f;
  for (int d = 0; d < HC; ++d) { float e = __expf(vals[d] - mx); vals[d] = e; sum += e; }
  const float inv = 1.f / sum;
  float* orow = attn + (size_t)bh * (HC * HC) + m * HC;
  for (int d = 0; d < HC; ++d) orow[d] = vals[d] * inv;
}

// ---------------------------------------------------------------------------
// Kernel 5: out = attn @ v.  M=48 (3 waves), K=48 padded to 64, N=16 pixels.
// v tile async-staged to LDS (96 x 16B chunks == one per thread).
// ---------------------------------------------------------------------------
__global__ __launch_bounds__(96) void av_kernel(
    const float* __restrict__ attn,   // [32][48][48]
    const __bf16* __restrict__ Q1,    // [B][576][NPIX]
    float* __restrict__ outA)         // [B][192][NPIX]
{
  __shared__ __bf16 tA[HC][64 + 8];   // [m][k], k 48..63 zero
  __shared__ __bf16 tV[64][16 + 8];   // [k=d][n], row stride 48B (16B aligned)

  const int p0 = blockIdx.x * 16;
  const int bh = blockIdx.y;
  const int b = bh >> 2, head = bh & 3;

  // async: 48 v-rows x 32B = 96 x 16B chunks, exactly one per thread
  const __bf16* vbase = Q1 + ((size_t)b * C3 + 2 * CDIM + head * HC) * NPIX + p0;
  {
    int dd = threadIdx.x >> 1, g = (threadIdx.x & 1) * 8;
    async_copy16(lds_off(&tV[dd][g]), vbase + (size_t)dd * NPIX + g);
  }
  // pad rows 48..63 with zeros
  for (int i = threadIdx.x; i < 16 * 16; i += 96) {
    tV[48 + (i >> 4)][i & 15] = f2bf(0.f);
  }
  // attn tile -> bf16 (K-padded)
  const float* arow = attn + (size_t)bh * (HC * HC);
  for (int i = threadIdx.x; i < HC * 64; i += 96) {
    int m = i >> 6, kk = i & 63;
    tA[m][kk] = f2bf(kk < HC ? arow[m * HC + kk] : 0.f);
  }
  async_wait_all();
  __syncthreads();

  const int wave = threadIdx.x >> 5;   // m-tile
  const int lane = threadIdx.x & 31;
  const int m0 = wave * 16;
  const int mrow = m0 + (lane & 15);
  const int n = lane & 15;
  const int kbA = (lane < 16) ? 0 : 16;
  const int kbB = (lane < 16) ? 0 : 16;

  v8f acc = {};
#pragma unroll
  for (int k0 = 0; k0 < 64; k0 += 32) {
    v16bf a, bb;
#pragma unroll
    for (int j = 0; j < 8; ++j) {
      a[j]     = tA[mrow][k0 + kbA + j];
      a[j + 8] = tA[mrow][k0 + kbA + 8 + j];
    }
#pragma unroll
    for (int j = 0; j < 16; ++j)
      bb[j] = tV[k0 + kbB + j][n];
    acc = __builtin_amdgcn_wmma_f32_16x16x32_bf16(false, a, false, bb,
                                                  (short)0, acc, false, false);
  }

  const int hi = lane >> 4;
  float* ob = outA + ((size_t)b * CDIM + head * HC) * NPIX;
#pragma unroll
  for (int r = 0; r < 8; ++r) {
    int orow = m0 + r + 8 * hi;
    ob[(size_t)orow * NPIX + p0 + n] = acc[r];
  }
}

// ---------------------------------------------------------------------------
extern "C" void kernel_launch(void* const* d_in, const int* in_sizes, int n_in,
                              void* d_out, int out_size, void* d_ws, size_t ws_size,
                              hipStream_t stream) {
  const float* x      = (const float*)d_in[0];   // [8,192,128,128]
  const float* w_qkv  = (const float*)d_in[1];   // [576,192]
  const float* w_dw   = (const float*)d_in[2];   // [576,1,3,3]
  const float* w_proj = (const float*)d_in[3];   // [192,192]
  const float* temp   = (const float*)d_in[4];   // [4]
  float* out = (float*)d_out;

  char* ws = (char*)d_ws;
  size_t off = 0;
  auto alloc = [&](size_t bytes) -> void* {
    void* p = ws + off;
    off = (off + bytes + 255) & ~(size_t)255;
    return p;
  };
  float*  qkv0      = (float*) alloc((size_t)BATCH * C3 * NPIX * sizeof(float));   // 302 MB
  __bf16* qkv1      = (__bf16*)alloc((size_t)BATCH * C3 * NPIX * sizeof(__bf16));  // 151 MB
  __bf16* wqkv_bf   = (__bf16*)alloc((size_t)C3 * CDIM * sizeof(__bf16));
  __bf16* wproj_bf  = (__bf16*)alloc((size_t)CDIM * CDIM * sizeof(__bf16));
  float*  sumsq_prt = (float*) alloc((size_t)BATCH * 384 * 64 * sizeof(float));
  float*  sumsq     = (float*) alloc((size_t)BATCH * 384 * sizeof(float));
  float*  attn_part = (float*) alloc((size_t)16 * 32 * HC * HC * sizeof(float));
  float*  attn      = (float*) alloc((size_t)32 * HC * HC * sizeof(float));
  float*  outA      = qkv0;   // qkv0 dead after dwconv; alias (100 MB <= 302 MB)

  // 0) one-shot weight conversion to bf16
  cvt_w_kernel<<<(C3 * CDIM + 255) / 256, 256, 0, stream>>>(w_qkv, wqkv_bf, C3 * CDIM);
  cvt_w_kernel<<<(CDIM * CDIM + 255) / 256, 256, 0, stream>>>(w_proj, wproj_bf, CDIM * CDIM);

  // 1) qkv = w_qkv @ x   (M=576, K=192)
  pw_gemm_kernel<<<dim3(NPIX / 16, C3 / 64, BATCH), 128, 0, stream>>>(
      wqkv_bf, x, qkv0, C3, CDIM);

  // 2) depthwise 3x3, f32 -> bf16, + q/k sum-of-squares partials
  dwconv_kernel<<<dim3(8, 8, BATCH * C3), 256, 0, stream>>>(
      qkv0, w_dw, qkv1, sumsq_prt);
  reduce_sumsq_kernel<<<(BATCH * 384 + 255) / 256, 256, 0, stream>>>(
      sumsq_prt, sumsq);

  // 3) Gram partials (K split 16 ways over pixels)
  gram_kernel<<<dim3(9, BATCH * HEADS, 16), 32, 0, stream>>>(qkv1, attn_part);

  // 4) reduce + norm-scale + temperature + softmax
  softmax_kernel<<<BATCH * HEADS, 64, 0, stream>>>(attn_part, sumsq, temp, attn);

  // 5) out = attn @ v
  av_kernel<<<dim3(NPIX / 16, BATCH * HEADS), 96, 0, stream>>>(attn, qkv1, outA);

  // 6) projection (M=192, K=192)
  pw_gemm_kernel<<<dim3(NPIX / 16, CDIM / 64, BATCH), 128, 0, stream>>>(
      wproj_bf, outA, out, CDIM, CDIM);
}
